// TransformerBlockQuantum_65481071403469
// MI455X (gfx1250) — matrix operand order repaired and checked by reading, not compiled, not run on hardware
//
#include <hip/hip_runtime.h>

#define BB 4
#define SS 2048
#define EE 512
#define HH 8
#define FF 2048

typedef __attribute__((ext_vector_type(16))) __bf16 v16bf;
typedef __attribute__((ext_vector_type(8)))  __bf16 v8bf;
typedef __attribute__((ext_vector_type(8)))  float  v8f;

__device__ __forceinline__ v8f wmma_bf16(v16bf a, v16bf b, v8f c) {
  return __builtin_amdgcn_wmma_f32_16x16x32_bf16(false, a, false, b, (short)0, c,
                                                 false, false);
}

// A fragment: 16(M) x 32(K) bf16 tile from row-major source (row stride ld elems).
// lanes 0-15 & 16-31 both hold rows M=0..15; lane<16 holds K = k0+0..7 (elems 0..7)
// and k0+16..23 (elems 8..15); lane>=16 holds k0+8..15 / k0+24..31.
__device__ __forceinline__ v16bf load_a_frag(const __bf16* base, int ld, int k0) {
  const int lane = threadIdx.x & 31;
  const int half = lane >> 4;
  const int row  = lane & 15;
  const __bf16* p = base + (size_t)row * ld + k0 + 8 * half;
  v8bf lo = *(const v8bf*)p;         // K = k0+8*half .. +7
  v8bf hi = *(const v8bf*)(p + 16);  // K = k0+16+8*half .. +7
  return __builtin_shufflevector(lo, hi, 0,1,2,3,4,5,6,7,8,9,10,11,12,13,14,15);
}

// B fragment: 32(K) x 16(N) from a source laid out T[n][k] (row n of T is column
// n0+n of B, contiguous in K). lanes 0-15: N=n0+lane, K=c0..c0+7, c0+16..c0+23;
// lanes 16-31: N=n0+lane-16, K=c0+8..c0+15, c0+24..c0+31.
__device__ __forceinline__ v16bf load_b_frag(const __bf16* t, int ld, int n0, int c0) {
  const int lane = threadIdx.x & 31;
  const __bf16* p = t + (size_t)(n0 + (lane & 15)) * ld + c0 + 8 * (lane >> 4);
  v8bf lo = *(const v8bf*)p;
  v8bf hi = *(const v8bf*)(p + 16);
  return __builtin_shufflevector(lo, hi, 0,1,2,3,4,5,6,7,8,9,10,11,12,13,14,15);
}

// ---------------- elementwise prep ----------------

__global__ __launch_bounds__(256) void qmap2_kernel(const float* __restrict__ x,
                                                    const float* __restrict__ th,
                                                    __bf16* __restrict__ xq, int n) {
  int i = blockIdx.x * 256 + threadIdx.x;
  if (i >= n) return;
  float c = __cosf(th[i & (EE - 1)]);
  float z = __cosf(x[i]) * c;        // first qmap
  xq[i] = (__bf16)(__cosf(z) * c);   // second qmap
}

__global__ __launch_bounds__(256) void f2bf_kernel(const float* __restrict__ in,
                                                   __bf16* __restrict__ out, int n) {
  int i = blockIdx.x * 256 + threadIdx.x;
  if (i < n) out[i] = (__bf16)in[i];
}

// ---------------- flash attention (q = k = v = xq head slice) ----------------
// grid: (SS/64, BB*HH), block 128 (4 waves). Each wave owns 16 query rows.

__global__ __launch_bounds__(128) void attn_kernel(const __bf16* __restrict__ xq,
                                                   float* __restrict__ attn_out) {
  const int tid  = threadIdx.x;
  const int wave = tid >> 5;
  const int lane = tid & 31;
  const int half = lane >> 4;
  const int lcol = lane & 15;
  const int b = blockIdx.y >> 3;   // / HH
  const int h = blockIdx.y & 7;    // % HH
  const int q0 = blockIdx.x * 64;

  __shared__ __align__(16) __bf16 vt[64][72];     // transposed V tile: vt[d][key]
  __shared__ __align__(16) __bf16 p_lds[64][72];  // probs (bf16)
  __shared__ __align__(16) float s_sc[64][68];    // scaled scores (16B-aligned rows)
  __shared__ float osc[64];                       // per-row rescale / final 1/l

  const __bf16* xq_bh = xq + ((size_t)b * SS) * EE + (size_t)h * 64;

  // Q fragments for this wave's 16 rows (K-dim = DK = 64 -> 2 chunks of 32)
  const __bf16* qbase = xq_bh + (size_t)(q0 + wave * 16) * EE;
  v16bf aq0 = load_a_frag(qbase, EE, 0);
  v16bf aq1 = load_a_frag(qbase, EE, 32);

  v8f o[4];
#pragma unroll
  for (int t = 0; t < 4; ++t)
#pragma unroll
    for (int r = 0; r < 8; ++r) o[t][r] = 0.f;

  float m_r = -1e30f, l_r = 0.f;  // live only in threads 0..63

  for (int jb = 0; jb < SS / 64; ++jb) {
    const int j0 = jb * 64;
    // stage transposed V tile (coalesced global read along d)
    for (int idx = tid; idx < 64 * 64; idx += 128) {
      int key = idx >> 6, d = idx & 63;
      vt[d][key] = xq_bh[(size_t)(j0 + key) * EE + d];
    }
    // prefetch next key tile (one cacheline probe per thread)
    if (j0 + 64 < SS)
      __builtin_prefetch(xq_bh + (size_t)(j0 + 64 + (tid >> 1)) * EE + (tid & 1) * 32,
                         0, 3);
    // scores S = Q*K^T * 1/sqrt(DK); B fragments come straight from global
    const __bf16* kbase = xq_bh + (size_t)j0 * EE;
#pragma unroll
    for (int t = 0; t < 4; ++t) {
      v8f acc;
#pragma unroll
      for (int r = 0; r < 8; ++r) acc[r] = 0.f;
      acc = wmma_bf16(aq0, load_b_frag(kbase, EE, t * 16, 0), acc);
      acc = wmma_bf16(aq1, load_b_frag(kbase, EE, t * 16, 32), acc);
#pragma unroll
      for (int r = 0; r < 8; ++r)
        s_sc[wave * 16 + r + 8 * half][t * 16 + lcol] = acc[r] * 0.125f;
    }
    __syncthreads();
    // online softmax: thread t (<64) owns score row t; vectorized row scan
    if (tid < 64) {
      const float4* prow = (const float4*)&s_sc[tid][0];
      float tm = -1e30f;
#pragma unroll
      for (int j = 0; j < 16; ++j) {
        float4 v = prow[j];
        tm = fmaxf(tm, fmaxf(fmaxf(v.x, v.y), fmaxf(v.z, v.w)));
      }
      float nm = fmaxf(m_r, tm);
      float sc = __expf(m_r - nm);
      float rs = 0.f;
#pragma unroll
      for (int j = 0; j < 16; ++j) {
        float4 v = prow[j];
        float p0 = __expf(v.x - nm), p1 = __expf(v.y - nm);
        float p2 = __expf(v.z - nm), p3 = __expf(v.w - nm);
        rs += (p0 + p1) + (p2 + p3);
        p_lds[tid][4 * j + 0] = (__bf16)p0;
        p_lds[tid][4 * j + 1] = (__bf16)p1;
        p_lds[tid][4 * j + 2] = (__bf16)p2;
        p_lds[tid][4 * j + 3] = (__bf16)p3;
      }
      l_r = l_r * sc + rs;
      m_r = nm;
      osc[tid] = sc;
    }
    __syncthreads();
    // rescale running O and accumulate O += P*V
#pragma unroll
    for (int t = 0; t < 4; ++t)
#pragma unroll
      for (int r = 0; r < 8; ++r) o[t][r] *= osc[wave * 16 + r + 8 * half];
    v16bf pa0 = load_a_frag(&p_lds[wave * 16][0], 72, 0);
    v16bf pa1 = load_a_frag(&p_lds[wave * 16][0], 72, 32);
#pragma unroll
    for (int t = 0; t < 4; ++t) {
      o[t] = wmma_bf16(pa0, load_b_frag(&vt[0][0], 72, t * 16, 0), o[t]);
      o[t] = wmma_bf16(pa1, load_b_frag(&vt[0][0], 72, t * 16, 32), o[t]);
    }
    __syncthreads();  // protect vt/p_lds/s_sc for next iteration
  }

  if (tid < 64) osc[tid] = 1.f / l_r;
  __syncthreads();

  float* obase = attn_out + ((size_t)b * SS + q0 + wave * 16) * EE + (size_t)h * 64;
#pragma unroll
  for (int t = 0; t < 4; ++t)
#pragma unroll
    for (int r = 0; r < 8; ++r) {
      int row = r + 8 * half;
      obase[(size_t)row * EE + t * 16 + lcol] = o[t][r] * osc[wave * 16 + row];
    }
}

// ---------------- layernorm (+ optional FFN qmap) ----------------
// one block per token row; E=512, 256 threads x 2 elems.

template <int WRITE_QF>
__global__ __launch_bounds__(256) void ln_kernel(const float* __restrict__ xa,
                                                 const float* __restrict__ xb,
                                                 const float* __restrict__ g,
                                                 const float* __restrict__ bt,
                                                 const float* __restrict__ th,
                                                 float* __restrict__ y,
                                                 __bf16* __restrict__ qf) {
  const int row = blockIdx.x;
  const int t = threadIdx.x;
  const float* pa = xa + (size_t)row * EE;
  const float* pb = xb + (size_t)row * EE;
  float v0 = pa[t] + pb[t];
  float v1 = pa[t + 256] + pb[t + 256];

  __shared__ float red[256];
  red[t] = v0 + v1;
  __syncthreads();
  for (int off = 128; off > 0; off >>= 1) {
    if (t < off) red[t] += red[t + off];
    __syncthreads();
  }
  float mean = red[0] * (1.f / EE);
  __syncthreads();
  float d0 = v0 - mean, d1 = v1 - mean;
  red[t] = d0 * d0 + d1 * d1;
  __syncthreads();
  for (int off = 128; off > 0; off >>= 1) {
    if (t < off) red[t] += red[t + off];
    __syncthreads();
  }
  float rsq = rsqrtf(red[0] * (1.f / EE) + 1e-5f);

  float y0 = d0 * rsq * g[t] + bt[t];
  float y1 = d1 * rsq * g[t + 256] + bt[t + 256];
  float* py = y + (size_t)row * EE;
  py[t] = y0;
  py[t + 256] = y1;
  if (WRITE_QF) {
    __bf16* pq = qf + (size_t)row * EE;
    pq[t]       = (__bf16)(__cosf(y0) * __cosf(th[t]));
    pq[t + 256] = (__bf16)(__cosf(y1) * __cosf(th[t + 256]));
  }
}

// ---------------- tiled bf16 GEMM: out = act(A @ W + bias) ----------------
// A: MxK bf16 row-major; W: KxN bf16 row-major.
// Block = 128 threads (4 waves), output tile 128x64, k-step 64.
// Each wave owns two 16-row subtiles; B fragments reused across both.

template <int RELU, int OUTBF>
__global__ __launch_bounds__(128) void gemm_kernel(const __bf16* __restrict__ A,
                                                   const __bf16* __restrict__ W,
                                                   const float* __restrict__ bias,
                                                   __bf16* __restrict__ outb,
                                                   float* __restrict__ outf,
                                                   int M, int N, int K) {
  const int tid  = threadIdx.x;
  const int wave = tid >> 5;
  const int lane = tid & 31;
  const int half = lane >> 4;
  const int lcol = lane & 15;
  const int n0 = blockIdx.x * 64;
  const int m0 = blockIdx.y * 128;

  __shared__ __align__(16) __bf16 wt[64][72];  // wt[n][k] = W[k0+k][n0+n]

  v8f acc[2][4];
#pragma unroll
  for (int rt = 0; rt < 2; ++rt)
#pragma unroll
    for (int t = 0; t < 4; ++t)
#pragma unroll
      for (int r = 0; r < 8; ++r) acc[rt][t][r] = 0.f;

  const __bf16* abase = A + (size_t)(m0 + wave * 32) * K;

  for (int k0 = 0; k0 < K; k0 += 64) {
    // stage 64(k) x 64(n) W tile, transposed into wt[n][k]
    for (int idx = tid; idx < 64 * 64; idx += 128) {
      int n = idx & 63, k = idx >> 6;
      wt[n][k] = W[(size_t)(k0 + k) * N + n0 + n];
    }
    // prefetch next W tile (one cacheline probe per thread)
    if (k0 + 64 < K)
      __builtin_prefetch(W + (size_t)(k0 + 64 + (tid >> 1)) * N + n0 + (tid & 1) * 32,
                         0, 3);
    __syncthreads();
    v16bf a00 = load_a_frag(abase, K, k0);
    v16bf a01 = load_a_frag(abase, K, k0 + 32);
    v16bf a10 = load_a_frag(abase + (size_t)16 * K, K, k0);
    v16bf a11 = load_a_frag(abase + (size_t)16 * K, K, k0 + 32);
#pragma unroll
    for (int t = 0; t < 4; ++t) {
      v16bf b0 = load_b_frag(&wt[0][0], 72, t * 16, 0);
      v16bf b1 = load_b_frag(&wt[0][0], 72, t * 16, 32);
      acc[0][t] = wmma_bf16(a00, b0, acc[0][t]);
      acc[0][t] = wmma_bf16(a01, b1, acc[0][t]);
      acc[1][t] = wmma_bf16(a10, b0, acc[1][t]);
      acc[1][t] = wmma_bf16(a11, b1, acc[1][t]);
    }
    __syncthreads();
  }

#pragma unroll
  for (int rt = 0; rt < 2; ++rt)
#pragma unroll
    for (int t = 0; t < 4; ++t) {
      int col = n0 + t * 16 + lcol;
      float bv = bias[col];
#pragma unroll
      for (int r = 0; r < 8; ++r) {
        size_t row = (size_t)(m0 + wave * 32 + rt * 16 + r + 8 * half);
        float v = acc[rt][t][r] + bv;
        if (RELU) v = fmaxf(v, 0.f);
        if (OUTBF) outb[row * N + col] = (__bf16)v;
        else       outf[row * N + col] = v;
      }
    }
}

// ---------------- launch ----------------

extern "C" void kernel_launch(void* const* d_in, const int* in_sizes, int n_in,
                              void* d_out, int out_size, void* d_ws, size_t ws_size,
                              hipStream_t stream) {
  const float* x       = (const float*)d_in[0];
  const float* th_attn = (const float*)d_in[1];
  const float* th_ffn  = (const float*)d_in[2];
  const float* W1      = (const float*)d_in[3];
  const float* b1      = (const float*)d_in[4];
  const float* W2      = (const float*)d_in[5];
  const float* b2      = (const float*)d_in[6];
  const float* g1      = (const float*)d_in[7];
  const float* bt1     = (const float*)d_in[8];
  const float* g2      = (const float*)d_in[9];
  const float* bt2     = (const float*)d_in[10];
  float* out = (float*)d_out;

  const size_t BSE = (size_t)BB * SS * EE;       // 4M tokens*feat
  const size_t MM  = (size_t)BB * SS;            // 8192 rows

  char* ws = (char*)d_ws;
  size_t off = 0;
  __bf16* xq   = (__bf16*)(ws + off); off += BSE * 2;
  float*  attn = (float*)(ws + off);  off += BSE * 4;
  float*  x1   = (float*)(ws + off);  off += BSE * 4;
  __bf16* qf   = (__bf16*)(ws + off); off += BSE * 2;
  __bf16* W1b  = (__bf16*)(ws + off); off += (size_t)EE * FF * 2;
  __bf16* W2b  = (__bf16*)(ws + off); off += (size_t)FF * EE * 2;
  __bf16* hbuf = (__bf16*)(ws + off); off += MM * FF * 2;
  float*  fbuf = (float*)(ws + off);  off += MM * EE * 4;
  (void)off; (void)ws_size; (void)in_sizes; (void)n_in; (void)out_size;

  // 1. xq = qmap(qmap(x)) in bf16
  qmap2_kernel<<<(int)(BSE / 256), 256, 0, stream>>>(x, th_attn, xq, (int)BSE);
  // 2. weights to bf16
  f2bf_kernel<<<(EE * FF) / 256, 256, 0, stream>>>(W1, W1b, EE * FF);
  f2bf_kernel<<<(FF * EE) / 256, 256, 0, stream>>>(W2, W2b, FF * EE);
  // 3. attention
  attn_kernel<<<dim3(SS / 64, BB * HH), 128, 0, stream>>>(xq, attn);
  // 4. x1 = LN(x + attn); qf = qmap_ffn(x1)
  ln_kernel<1><<<(int)MM, 256, 0, stream>>>(x, attn, g1, bt1, th_ffn, x1, qf);
  // 5. h = relu(qf @ W1 + b1)  (bf16 out)
  gemm_kernel<1, 1><<<dim3(FF / 64, (int)(MM / 128)), 128, 0, stream>>>(
      qf, W1b, b1, hbuf, nullptr, (int)MM, FF, EE);
  // 6. f = h @ W2 + b2  (f32 out)
  gemm_kernel<0, 0><<<dim3(EE / 64, (int)(MM / 128)), 128, 0, stream>>>(
      hbuf, W2b, b2, nullptr, fbuf, (int)MM, EE, FF);
  // 7. out = LN(x1 + f)
  ln_kernel<0><<<(int)MM, 256, 0, stream>>>(x1, fbuf, g2, bt2, nullptr, out, nullptr);
}
